// DLRMV2_40604620817052
// MI455X (gfx1250) — matrix operand order repaired and checked by reading, not compiled, not run on hardware
//
#include <hip/hip_runtime.h>
#include <hip/hip_bf16.h>

#define BATCH_N 16384
#define EMB_DIM 128
#define N_TABLES 26
#define TOP_IN 3456   // 128 + 26*128

typedef unsigned short u16;
typedef __attribute__((ext_vector_type(16))) __bf16 v16bf;
typedef __attribute__((ext_vector_type(8)))  float  v8f;
typedef __attribute__((ext_vector_type(4))) unsigned int u32x4;
typedef __attribute__((ext_vector_type(8))) int i32x8;
typedef __attribute__((ext_vector_type(4))) int i32x4;

#ifndef HAVE_TDM
#if defined(__has_builtin)
#if __has_builtin(__builtin_amdgcn_tensor_load_to_lds) && __has_builtin(__builtin_amdgcn_s_wait_tensorcnt)
#define HAVE_TDM 1
#endif
#endif
#endif
#ifndef HAVE_TDM
#define HAVE_TDM 0
#endif

__device__ __forceinline__ u16 f2bf(float f) {
  unsigned u = __builtin_bit_cast(unsigned, f);
  u += 0x7FFFu + ((u >> 16) & 1u);     // round-to-nearest-even
  return (u16)(u >> 16);
}
__device__ __forceinline__ float bf2f(u16 h) {
  unsigned u = ((unsigned)h) << 16;
  return __builtin_bit_cast(float, u);
}

union FragU { struct { uint4 lo, hi; } q; v16bf v; };

// LDS slab geometry: 128 rows x 32 bf16, padded to 40 el (80B) rows -> conflict-free b128
#define SLAB_EL   (128 * 40)
#define SLAB_BY   (SLAB_EL * 2)

#if HAVE_TDM
// Issue a TDM 2D tile load: 32 (K) x 128 (rows) bf16 tile from row-major [rows x K]
// global matrix into LDS at byte offset lds_off, padding +16B per 64B row.
__device__ __forceinline__ void tdm_load_2d(unsigned lds_off, const u16* gsrc,
                                            int K, int rows) {
  unsigned long long ga = (unsigned long long)(__UINTPTR_TYPE__)gsrc;
  u32x4 g0;
  g0[0] = 1u;                                          // count=1, user mode
  g0[1] = lds_off;                                     // lds_addr (bytes)
  g0[2] = (unsigned)(ga & 0xFFFFFFFFull);              // global_addr[31:0]
  g0[3] = (unsigned)((ga >> 32) & 0x1FFFFFFull) | (2u << 30);  // addr[56:32] | type=2
  i32x8 g1;
  // data_size=1 (2B), pad_enable, pad_interval=3 (16 dwords), pad_amount=3 (4 dwords)
  g1[0] = (int)((1u << 16) | (1u << 20) | (3u << 22) | (3u << 25));
  g1[1] = (int)(((unsigned)K & 0xFFFFu) << 16);                         // tensor_dim0 lo
  g1[2] = (int)(((unsigned)K >> 16) | (((unsigned)rows & 0xFFFFu) << 16)); // dim0 hi | dim1 lo
  g1[3] = (int)(((unsigned)rows >> 16) | (32u << 16));                  // dim1 hi | tile_dim0=32
  g1[4] = (int)128u;                                                    // tile_dim1=128, tile_dim2=0
  g1[5] = (int)K;                                                       // tensor_dim0_stride lo
  g1[6] = 0;
  g1[7] = 0;
  i32x4 z4 = {0, 0, 0, 0};
#if __clang_major__ >= 23
  i32x8 z8 = {0, 0, 0, 0, 0, 0, 0, 0};
  __builtin_amdgcn_tensor_load_to_lds(g0, g1, z4, z4, z8, 0);
#else
  __builtin_amdgcn_tensor_load_to_lds(g0, g1, z4, z4, 0);
#endif
}
#endif

// ---------------- bottom layer 1 : [B,13] x [13,512] + b, ReLU -> bf16 ----------
__global__ void k_bottom_l1(const float* __restrict__ X, const float* __restrict__ W,
                            const float* __restrict__ b, u16* __restrict__ out) {
  __shared__ float sW[13 * 512];
  for (int i = threadIdx.x; i < 13 * 512; i += blockDim.x) sW[i] = W[i];
  __syncthreads();
  int idx = blockIdx.x * blockDim.x + threadIdx.x;   // over B*512 exactly
  int row = idx >> 9, col = idx & 511;
  float s = b[col];
  const float* xr = X + row * 13;
#pragma unroll
  for (int k = 0; k < 13; ++k) s = fmaf(xr[k], sW[k * 512 + col], s);
  out[idx] = f2bf(fmaxf(s, 0.0f));
}

// ------------- weight convert + transpose: W[K,N] f32 -> Wt[N,K] bf16 ------------
__global__ void k_cvt_wT(const float* __restrict__ W, u16* __restrict__ Wt, int K, int N) {
  int idx = blockIdx.x * blockDim.x + threadIdx.x;
  if (idx >= K * N) return;
  int k = idx / N, n = idx - k * N;
  Wt[(size_t)n * K + k] = f2bf(W[idx]);
}

// ---------------- gather + concat -> Xcat bf16 [B, 3456] -------------------------
struct EmbPtrs { const float* p[N_TABLES]; };

__global__ void k_gather_concat(EmbPtrs tabs, const int* __restrict__ ids,
                                const u16* __restrict__ xbot, u16* __restrict__ Xcat) {
  int b   = blockIdx.x;
  int seg = blockIdx.y;            // 0 = bottom-MLP output, 1..26 = tables
  int t   = threadIdx.x;           // 0..127
  u16* dst = Xcat + (size_t)b * TOP_IN + seg * EMB_DIM + t;
  if (seg == 0) {
    *dst = xbot[(size_t)b * EMB_DIM + t];
  } else {
    int tab = seg - 1;
    int id  = ids[tab * BATCH_N + b];
    *dst = f2bf(tabs.p[tab][(size_t)id * EMB_DIM + t]);
  }
}

// ---- LDS-tiled bf16 WMMA GEMM: C[M,N] = act(A[M,K] x W + bias), W stored [N,K] ---
// Block: 256 threads (8 waves) computes a 128x128 tile of C; wave w -> 32x64 subtile.
// K-slabs of 32 double-buffered in LDS, staged by the Tensor Data Mover.
__global__ void k_gemm_tile(const u16* __restrict__ A, const u16* __restrict__ Wt,
                            const float* __restrict__ bias, u16* __restrict__ C,
                            int M, int N, int K, int relu) {
  extern __shared__ u16 sm[];            // [A0 | A1 | B0 | B1], each SLAB_EL
  const int tid  = threadIdx.x;
  const int lane = tid & 31;
  const int w    = tid >> 5;             // 0..7
  const int half = lane >> 4;            // lane 0-15 / 16-31
  const int r    = lane & 15;
  const int nbn  = N >> 7;
  const int bM   = (blockIdx.x / nbn) << 7;
  const int bN   = (blockIdx.x % nbn) << 7;
  const int wm   = (w >> 1) << 5;        // 0,32,64,96 (M offset in block)
  const int wn   = (w & 1) << 6;         // 0,64       (N offset in block)
  const int nk   = K >> 5;

  v8f acc[2][4] = {};

#if HAVE_TDM
  if (tid < 32) {                         // wave 0 drives the TDM
    tdm_load_2d(0u,           A  + (size_t)bM * K, K, M);
    tdm_load_2d(2u * SLAB_BY, Wt + (size_t)bN * K, K, N);
  }
#endif

  for (int kt = 0; kt < nk; ++kt) {
    const int cur = kt & 1;
    u16* sAc = sm + cur * SLAB_EL;                 // A slab for this iteration
    u16* sBc = sm + 2 * SLAB_EL + cur * SLAB_EL;   // B slab for this iteration
#if HAVE_TDM
    if (tid < 32) {
      if (kt + 1 < nk) {
        const int nxt = (kt + 1) & 1;
        tdm_load_2d((unsigned)(nxt * SLAB_BY),
                    A + (size_t)bM * K + (size_t)(kt + 1) * 32, K, M);
        tdm_load_2d((unsigned)(2 * SLAB_BY + nxt * SLAB_BY),
                    Wt + (size_t)bN * K + (size_t)(kt + 1) * 32, K, N);
        __builtin_amdgcn_s_wait_tensorcnt(2);   // slab kt complete (in-order)
      } else {
        __builtin_amdgcn_s_wait_tensorcnt(0);
      }
    }
#else
    // cooperative staging fallback: 512 x 16B per slab across 256 threads
    for (int c = tid; c < 512; c += 256) {
      int row = c >> 2, q = (c & 3) * 8;
      *(uint4*)(sAc + row * 40 + q) =
          *(const uint4*)(A + (size_t)(bM + row) * K + (size_t)kt * 32 + q);
      *(uint4*)(sBc + row * 40 + q) =
          *(const uint4*)(Wt + (size_t)(bN + row) * K + (size_t)kt * 32 + q);
    }
#endif
    __syncthreads();                      // slab kt visible to all waves

    FragU a[2], b[4];
#pragma unroll
    for (int i = 0; i < 2; ++i) {
      const u16* p = sAc + (wm + i * 16 + r) * 40 + half * 8;
      a[i].q.lo = *(const uint4*)(p);        // K = kb..kb+7
      a[i].q.hi = *(const uint4*)(p + 16);   // K = kb+16..kb+23
    }
#pragma unroll
    for (int j = 0; j < 4; ++j) {
      const u16* p = sBc + (wn + j * 16 + r) * 40 + half * 8;
      b[j].q.lo = *(const uint4*)(p);        // K = kb..kb+7
      b[j].q.hi = *(const uint4*)(p + 16);   // K = kb+16..kb+23
    }
#pragma unroll
    for (int i = 0; i < 2; ++i)
#pragma unroll
      for (int j = 0; j < 4; ++j)
        acc[i][j] = __builtin_amdgcn_wmma_f32_16x16x32_bf16(
            false, a[i].v, false, b[j].v, (short)0, acc[i][j], false, false);

    __syncthreads();                      // everyone done reading before overwrite
  }

#pragma unroll
  for (int i = 0; i < 2; ++i) {
#pragma unroll
    for (int j = 0; j < 4; ++j) {
      const int col = bN + wn + j * 16 + r;
      const float bb = bias[col];
#pragma unroll
      for (int e = 0; e < 8; ++e) {
        float v = acc[i][j][e] + bb;
        if (relu) v = fmaxf(v, 0.0f);
        // C/D layout: VGPR e -> row +e (lanes 0-15) / +e+8 (lanes 16-31)
        C[(size_t)(bM + wm + i * 16 + e + half * 8) * N + col] = f2bf(v);
      }
    }
  }
}

// ---------------- final layer: [B,256] . w[256] + b0 -> float ---------------------
__global__ void k_final_dot(const u16* __restrict__ A, const u16* __restrict__ w,
                            const float* __restrict__ bias, float* __restrict__ out) {
  int lane = threadIdx.x & 31;
  int row  = blockIdx.x * (blockDim.x >> 5) + (threadIdx.x >> 5);
  uint4 av = *(const uint4*)(A + (size_t)row * 256 + lane * 8);
  uint4 wv = *(const uint4*)(w + lane * 8);
  const u16* ap = (const u16*)&av;
  const u16* wp = (const u16*)&wv;
  float s = 0.f;
#pragma unroll
  for (int j = 0; j < 8; ++j) s = fmaf(bf2f(ap[j]), bf2f(wp[j]), s);
#pragma unroll
  for (int off = 16; off; off >>= 1) s += __shfl_down(s, off, 32);
  if (lane == 0) out[row] = s + bias[0];
}

extern "C" void kernel_launch(void* const* d_in, const int* in_sizes, int n_in,
                              void* d_out, int out_size, void* d_ws, size_t ws_size,
                              hipStream_t stream) {
  const int B = BATCH_N;
  const float* dense = (const float*)d_in[0];
  const float* Wb1 = (const float*)d_in[1];
  const float* Wb2 = (const float*)d_in[2];
  const float* Wb3 = (const float*)d_in[3];
  const float* bb1 = (const float*)d_in[4];
  const float* bb2 = (const float*)d_in[5];
  const float* bb3 = (const float*)d_in[6];
  EmbPtrs tabs;
  for (int i = 0; i < N_TABLES; ++i) tabs.p[i] = (const float*)d_in[7 + i];
  const float* WT[5]; const float* bT[5];
  for (int i = 0; i < 5; ++i) { WT[i] = (const float*)d_in[33 + i]; bT[i] = (const float*)d_in[38 + i]; }
  const int* ids = (const int*)d_in[43];

  // --- carve bf16 workspace out of d_ws ---
  char* ws = (char*)d_ws;
  size_t off = 0;
  auto alloc = [&](size_t elems) -> u16* {
    u16* p = (u16*)(ws + off);
    off += ((elems * sizeof(u16) + 255) & ~(size_t)255);
    return p;
  };
  u16* WtB2 = alloc(512 * 256);
  u16* WtB3 = alloc(256 * 128);
  u16* WtT1 = alloc((size_t)TOP_IN * 1024);
  u16* WtT2 = alloc(1024 * 1024);
  u16* WtT3 = alloc(1024 * 512);
  u16* WtT4 = alloc(512 * 256);
  u16* wT5  = alloc(256);
  u16* h1   = alloc((size_t)B * 512);
  u16* h2   = alloc((size_t)B * 256);
  u16* h3   = alloc((size_t)B * 128);
  u16* Xcat = alloc((size_t)B * TOP_IN);
  u16* t1   = alloc((size_t)B * 1024);
  u16* t2   = alloc((size_t)B * 1024);
  u16* t3   = alloc((size_t)B * 512);
  u16* t4   = alloc((size_t)B * 256);
  (void)ws_size; (void)in_sizes; (void)n_in; (void)out_size;

  // --- weight conversions (independent, stream-ordered before first use) ---
  auto cvt = [&](const float* W, u16* Wt, int K, int N) {
    int n = K * N;
    k_cvt_wT<<<(n + 255) / 256, 256, 0, stream>>>(W, Wt, K, N);
  };
  cvt(Wb2, WtB2, 512, 256);
  cvt(Wb3, WtB3, 256, 128);
  cvt(WT[0], WtT1, TOP_IN, 1024);
  cvt(WT[1], WtT2, 1024, 1024);
  cvt(WT[2], WtT3, 1024, 512);
  cvt(WT[3], WtT4, 512, 256);
  cvt(WT[4], wT5, 256, 1);

  // --- bottom MLP ---
  k_bottom_l1<<<(B * 512) / 256, 256, 0, stream>>>(dense, Wb1, bb1, h1);
  auto gemm = [&](const u16* A, const u16* Wt, const float* bias, u16* C,
                  int M, int N, int K, int relu) {
    int blocks = (M >> 7) * (N >> 7);
    k_gemm_tile<<<blocks, 256, 4 * SLAB_BY, stream>>>(A, Wt, bias, C, M, N, K, relu);
  };
  gemm(h1, WtB2, bb2, h2, B, 256, 512, 1);
  gemm(h2, WtB3, bb3, h3, B, 128, 256, 1);

  // --- embedding gather + concat ---
  dim3 gg(B, N_TABLES + 1);
  k_gather_concat<<<gg, EMB_DIM, 0, stream>>>(tabs, ids, h3, Xcat);

  // --- top MLP ---
  gemm(Xcat, WtT1, bT[0], t1, B, 1024, TOP_IN, 1);
  gemm(t1,   WtT2, bT[1], t2, B, 1024, 1024, 1);
  gemm(t2,   WtT3, bT[2], t3, B, 512, 1024, 1);
  gemm(t3,   WtT4, bT[3], t4, B, 256, 512, 1);
  k_final_dot<<<B / 8, 256, 0, stream>>>(t4, wT5, bT[4], (float*)d_out);
}
